// AdaptiveLSTMBlockWrapper_48043504173414
// MI455X (gfx1250) — matrix operand order repaired and checked by reading, not compiled, not run on hardware
//
#include <hip/hip_runtime.h>
#include <hip/hip_bf16.h>
#include <cmath>

// ---------------------------------------------------------------------------
// AdaptiveLSTMBlock (ACT ponder LSTM) for MI455X / gfx1250.
//  B=4096, IN=1024, H=1024, T=12.
//  bf16 WMMA (v_wmma_f32_16x16x32_bf16) GEMMs, f32 accumulate.
//  CDNA5 data path: GLOBAL_LOAD_ASYNC_TO_LDS_B128 (ASYNCcnt) with
//  double-buffered LDS -> one s_wait_asynccnt + one barrier per K-step.
//  All 6 fragments preloaded per K-step so the 8 WMMAs issue as one burst.
//  Workspace requirement: ~200 MB.
// ---------------------------------------------------------------------------

constexpr int   kB   = 4096;
constexpr int   kIN  = 1024;
constexpr int   kH   = 1024;
constexpr int   kT   = 12;
constexpr int   k4H  = 4 * kH;
constexpr float kBudget      = 1.0f - 0.01f;   // 1 - PONDER_EPS
constexpr float kTimePenalty = 0.01f;

typedef __attribute__((ext_vector_type(16))) __bf16 v16bf;
typedef __attribute__((ext_vector_type(8)))  float  v8f;

__device__ __forceinline__ float sigmoidf_(float x) {
  return 1.0f / (1.0f + expf(-x));
}

// Direct global->LDS async copy (CDNA5). dst_lds: byte offset inside LDS
// (low 32 bits of a generic pointer to __shared__), src: global address.
__device__ __forceinline__ void async_copy_b128(void* dst_lds, const void* src)
{
  const unsigned lds_off = (unsigned)(unsigned long long)dst_lds;
  asm volatile("global_load_async_to_lds_b128 %0, %1, off"
               :: "v"(lds_off), "v"(src)
               : "memory");
}

__device__ __forceinline__ void wait_async_all()
{
  asm volatile("s_wait_asynccnt 0" ::: "memory");
}

// ---------------------------------------------------------------------------
// bf16 WMMA GEMM:  out[M,N] = A[M,K] * W[N,K]^T (+ addend[M,N]) (+ bias[N])
// Tile: 128x128x32, 256 threads = 8 waves (2 M x 4 N), wave tile 64x32,
// 4x2 accumulators of v_wmma_f32_16x16x32_bf16.
// Double-buffered LDS filled by global_load_async_to_lds_b128.
// ---------------------------------------------------------------------------
constexpr int TBM = 128, TBN = 128, TBK = 32;
constexpr int LDSS = TBK + 8;   // bf16 elems; 80B row stride, keeps 16B align

__global__ __launch_bounds__(256)
void gemm_bf16_wmma(float* __restrict__ out,
                    const __bf16* __restrict__ A,
                    const __bf16* __restrict__ W,
                    const float* __restrict__ addend,
                    const float* __restrict__ bias,
                    int M, int N, int K,
                    const int* __restrict__ active)
{
  if (active && *active == 0) return;

  __shared__ __bf16 As[2][TBM * LDSS];
  __shared__ __bf16 Ws[2][TBN * LDSS];

  const int tid  = threadIdx.x;
  const int lane = tid & 31;
  const int wave = tid >> 5;     // 0..7
  const int wm   = wave >> 2;    // 0..1 (M dir)
  const int wn   = wave & 3;     // 0..3 (N dir)
  const long row0 = (long)blockIdx.y * TBM;
  const long col0 = (long)blockIdx.x * TBN;

  const int fr  = lane & 15;     // fragment row (M for A, N for B)
  const int hi  = lane >> 4;     // lane half select
  const int aK0 = hi * 8;        // A frag: K chunks [aK0..aK0+7],[aK0+16..+23]
  const int bK0 = hi * 8;        // B frag: K chunks [bK0..bK0+7],[bK0+16..+23]

  // Per-thread staging coordinates: tile = 128x32 bf16 = 512 uint4;
  // 256 threads issue 2 async b128 copies each for A and for W.
  const int r0 = tid >> 2;                  // row for chunk q=0
  const int c0 = (tid & 3) << 3;            // bf16 col (x8)
  const int r1 = (tid + 256) >> 2;          // row for chunk q=1
  const int c1 = c0;                        // same column pattern

  v8f acc[4][2];
#pragma unroll
  for (int i = 0; i < 4; ++i)
#pragma unroll
    for (int j = 0; j < 2; ++j)
#pragma unroll
      for (int e = 0; e < 8; ++e) acc[i][j][e] = 0.0f;

  union Frag { v16bf v; uint4 u[2]; };

  auto issue_tile = [&](int buf, int k0) {
    async_copy_b128(&As[buf][r0 * LDSS + c0], &A[(row0 + r0) * (long)K + k0 + c0]);
    async_copy_b128(&Ws[buf][r0 * LDSS + c0], &W[(col0 + r0) * (long)K + k0 + c0]);
    async_copy_b128(&As[buf][r1 * LDSS + c1], &A[(row0 + r1) * (long)K + k0 + c1]);
    async_copy_b128(&Ws[buf][r1 * LDSS + c1], &W[(col0 + r1) * (long)K + k0 + c1]);
  };

  issue_tile(0, 0);
  int buf = 0;

  for (int k0 = 0; k0 < K; k0 += TBK) {
    // Wave's own async copies done, then barrier => whole tile visible, and
    // every wave has finished reading the *other* buffer (its ds_loads were
    // consumed by WMMAs before it could signal).
    wait_async_all();
    __syncthreads();
    if (k0 + TBK < K) issue_tile(buf ^ 1, k0 + TBK);

    // Preload ALL fragments into distinct registers so the 8 WMMAs below can
    // issue back-to-back (single dscnt wait, no per-pair stalls).
    Frag bf[2];
#pragma unroll
    for (int tn = 0; tn < 2; ++tn) {
      const __bf16* s = &Ws[buf][(wn * 32 + tn * 16 + fr) * LDSS + bK0];
      bf[tn].u[0] = *reinterpret_cast<const uint4*>(s);
      bf[tn].u[1] = *reinterpret_cast<const uint4*>(s + 16);
    }
    Frag af[4];
#pragma unroll
    for (int tm = 0; tm < 4; ++tm) {
      const __bf16* s = &As[buf][(wm * 64 + tm * 16 + fr) * LDSS];
      af[tm].u[0] = *reinterpret_cast<const uint4*>(s + aK0);
      af[tm].u[1] = *reinterpret_cast<const uint4*>(s + aK0 + 16);
    }
#pragma unroll
    for (int tm = 0; tm < 4; ++tm) {
#pragma unroll
      for (int tn = 0; tn < 2; ++tn) {
        acc[tm][tn] = __builtin_amdgcn_wmma_f32_16x16x32_bf16(
            false, af[tm].v, false, bf[tn].v, (short)0, acc[tm][tn], false, false);
      }
    }
    buf ^= 1;
  }

  // Epilogue: C layout — VGPR v holds M = v + 8*hi, lane%16 = N.
#pragma unroll
  for (int tm = 0; tm < 4; ++tm) {
#pragma unroll
    for (int tn = 0; tn < 2; ++tn) {
      const long col = col0 + wn * 32 + tn * 16 + fr;
      const float bv = bias ? bias[col] : 0.0f;
#pragma unroll
      for (int v = 0; v < 8; ++v) {
        const long row = row0 + wm * 64 + tm * 16 + hi * 8 + v;
        const long off = row * (long)N + col;
        float x = acc[tm][tn][v] + bv;
        if (addend) x += addend[off];
        out[off] = x;
      }
    }
  }
}

// ---------------------------------------------------------------------------
// Elementwise LSTM cell: gates [B,4H] (i,f,g,o), updates c,h (f32) and emits
// bf16 copies of h (for next hh-GEMM) and optionally relu(h) (layer-2 input).
// ---------------------------------------------------------------------------
__global__ __launch_bounds__(256)
void lstm_cell_kernel(const float* __restrict__ gates,
                      float* __restrict__ c_state,
                      float* __restrict__ h_state,
                      __bf16* __restrict__ h_bf,
                      __bf16* __restrict__ act_bf,
                      const int* __restrict__ active)
{
  if (*active == 0) return;
  const long idx = (long)blockIdx.x * blockDim.x + threadIdx.x;
  if (idx >= (long)kB * kH) return;
  const long b = idx / kH;
  const long j = idx - b * kH;
  const float* g = gates + b * (long)k4H + j;
  const float ig = sigmoidf_(g[0]);
  const float fg = sigmoidf_(g[kH]);
  const float gg = tanhf(g[2 * kH]);
  const float og = sigmoidf_(g[3 * kH]);
  const float c  = fg * c_state[idx] + ig * gg;
  const float h  = og * tanhf(c);
  c_state[idx] = c;
  h_state[idx] = h;
  h_bf[idx] = (__bf16)h;
  if (act_bf) act_bf[idx] = (__bf16)fmaxf(h, 0.0f);
}

// ---------------------------------------------------------------------------
// Halting GEMV + ACT bookkeeping. One block per sample.
// ---------------------------------------------------------------------------
__global__ __launch_bounds__(256)
void halt_kernel(const float* __restrict__ h2,
                 const float* __restrict__ W_halt,
                 const float* __restrict__ b_halt,
                 float* __restrict__ halt_accum,
                 int*   __restrict__ cont,
                 float* __restrict__ acc_rem,
                 float* __restrict__ acc_out,
                 float* __restrict__ ponder_steps,
                 const int* __restrict__ active_cur,
                 int*   __restrict__ active_next)
{
  if (*active_cur == 0) return;
  const int b = blockIdx.x;
  __shared__ float red[256];
  __shared__ float coefS;

  const float* hrow = h2 + (long)b * kH;
  float s = 0.0f;
  for (int j = threadIdx.x; j < kH; j += 256) s += hrow[j] * W_halt[j];
  red[threadIdx.x] = s;
  __syncthreads();
#pragma unroll
  for (int off = 128; off > 0; off >>= 1) {
    if (threadIdx.x < off) red[threadIdx.x] += red[threadIdx.x + off];
    __syncthreads();
  }
  if (threadIdx.x == 0) {
    const float step_halt = sigmoidf_(red[0] + b_halt[0]);
    const int   c  = cont[b];
    const float cf = c ? 1.0f : 0.0f;
    const float masked_halt = step_halt * cf;
    const float halt_n = halt_accum[b] + masked_halt;
    const int   ending = (halt_n > kBudget) && c;
    const int   cont_n = c && !ending;
    const float masked_rem = ending ? (1.0f - halt_n) : 0.0f;
    halt_accum[b]   = halt_n;
    acc_rem[b]     += masked_halt;
    ponder_steps[b] += cont_n ? 1.0f : 0.0f;
    cont[b] = cont_n;
    if (cont_n) atomicAdd(active_next, 1);
    coefS = masked_halt + masked_rem;
  }
  __syncthreads();
  const float w = coefS;
  if (w != 0.0f) {
    for (int j = threadIdx.x; j < kH; j += 256)
      acc_out[(long)b * kH + j] += w * hrow[j];
  }
}

// ---------------------------------------------------------------------------
// Finalization: remainder for still-continuing samples, ponder-cost reduce.
// ---------------------------------------------------------------------------
__global__ __launch_bounds__(256)
void final_acc_kernel(float* __restrict__ acc_out,
                      const float* __restrict__ h2,
                      const int* __restrict__ cont,
                      const float* __restrict__ halt_accum)
{
  const long idx = (long)blockIdx.x * blockDim.x + threadIdx.x;
  if (idx >= (long)kB * kH) return;
  const long b = idx / kH;
  if (cont[b]) acc_out[idx] += (1.0f - halt_accum[b]) * h2[idx];
}

__global__ __launch_bounds__(256)
void ponder_cost_kernel(const float* __restrict__ acc_rem, float* __restrict__ out)
{
  __shared__ float red[256];
  float s = 0.0f;
  for (int i = threadIdx.x; i < kB; i += 256) s += acc_rem[i];
  red[threadIdx.x] = s;
  __syncthreads();
#pragma unroll
  for (int off = 128; off > 0; off >>= 1) {
    if (threadIdx.x < off) red[threadIdx.x] += red[threadIdx.x + off];
    __syncthreads();
  }
  if (threadIdx.x == 0) *out = -kTimePenalty * (red[0] / (float)kB);
}

// ---------------------------------------------------------------------------
// Init: zero output states + per-sample ACT state + active counters.
// ---------------------------------------------------------------------------
__global__ __launch_bounds__(256)
void init_kernel(float* __restrict__ outbuf,
                 __bf16* __restrict__ h1_bf, __bf16* __restrict__ h2_bf,
                 float* __restrict__ halt_accum, float* __restrict__ acc_rem,
                 int* __restrict__ cont, int* __restrict__ active)
{
  const size_t idx = (size_t)blockIdx.x * blockDim.x + threadIdx.x;
  const size_t BH = (size_t)kB * kH;
  if (idx < 5 * BH) outbuf[idx] = 0.0f;                 // acc_out,h1,c1,h2,c2
  if (idx < BH) { h1_bf[idx] = (__bf16)0.0f; h2_bf[idx] = (__bf16)0.0f; }
  if (idx < (size_t)kB) {
    halt_accum[idx] = 0.0f;
    acc_rem[idx]    = 0.0f;
    cont[idx]       = 1;
    outbuf[5 * BH + 1 + idx] = 0.0f;                    // ponder_steps
  }
  if (idx == 0) {
    outbuf[5 * BH] = 0.0f;                              // ponder_cost
    active[0] = kB;
    for (int t = 1; t <= kT; ++t) active[t] = 0;
  }
}

// ---------------------------------------------------------------------------
// Weight prep: bf16 conversions; W_ih1 split into [4H,IN] main + flag column;
// fused biases.
// ---------------------------------------------------------------------------
__global__ __launch_bounds__(256)
void f32_to_bf16_kernel(const float* __restrict__ src, __bf16* __restrict__ dst, long n)
{
  const long i = (long)blockIdx.x * blockDim.x + threadIdx.x;
  if (i < n) dst[i] = (__bf16)src[i];
}

__global__ __launch_bounds__(256)
void prep_weights_kernel(const float* __restrict__ W_ih1,
                         __bf16* __restrict__ Wmain, float* __restrict__ flagcol,
                         const float* __restrict__ b_ih1, const float* __restrict__ b_hh1,
                         float* __restrict__ bias1,
                         const float* __restrict__ b_ih2, const float* __restrict__ b_hh2,
                         float* __restrict__ bias2)
{
  const long i = (long)blockIdx.x * blockDim.x + threadIdx.x;
  if (i < (long)k4H * kIN) {
    const long n = i / kIN;
    const long k = i - n * kIN;
    Wmain[i] = (__bf16)W_ih1[n * (kIN + 1) + k];
  }
  if (i < k4H) {
    flagcol[i] = W_ih1[i * (long)(kIN + 1) + kIN];
    bias1[i]   = b_ih1[i] + b_hh1[i];
    bias2[i]   = b_ih2[i] + b_hh2[i];
  }
}

// ---------------------------------------------------------------------------
// Host launch
// ---------------------------------------------------------------------------
static inline void launch_gemm(float* out, const __bf16* A, const __bf16* W,
                               const float* addend, const float* bias,
                               int M, int N, int K, const int* active,
                               hipStream_t stream)
{
  dim3 grid(N / TBN, M / TBM), block(256);
  gemm_bf16_wmma<<<grid, block, 0, stream>>>(out, A, W, addend, bias, M, N, K, active);
}

extern "C" void kernel_launch(void* const* d_in, const int* in_sizes, int n_in,
                              void* d_out, int out_size, void* d_ws, size_t ws_size,
                              hipStream_t stream)
{
  (void)in_sizes; (void)n_in; (void)out_size; (void)ws_size;

  const float* inputs = (const float*)d_in[0];
  const float* W_ih1  = (const float*)d_in[1];
  const float* W_hh1  = (const float*)d_in[2];
  const float* b_ih1  = (const float*)d_in[3];
  const float* b_hh1  = (const float*)d_in[4];
  const float* W_ih2  = (const float*)d_in[5];
  const float* W_hh2  = (const float*)d_in[6];
  const float* b_ih2  = (const float*)d_in[7];
  const float* b_hh2  = (const float*)d_in[8];
  const float* W_halt = (const float*)d_in[9];
  const float* b_halt = (const float*)d_in[10];

  const size_t BH = (size_t)kB * kH;
  float* outf         = (float*)d_out;
  float* acc_out      = outf;                 // [B,H]
  float* h1           = outf + 1 * BH;        // [B,H]
  float* c1           = outf + 2 * BH;        // [B,H]
  float* h2           = outf + 3 * BH;        // [B,H]
  float* c2           = outf + 4 * BH;        // [B,H]
  float* ponder_cost  = outf + 5 * BH;        // [1]
  float* ponder_steps = ponder_cost + 1;      // [B]

  // Workspace carve-up (256B aligned slices)
  char* p = (char*)d_ws;
  auto carve = [&](size_t bytes) -> char* {
    char* r = p; p += (bytes + 255) & ~(size_t)255; return r;
  };
  __bf16* Wih1b = (__bf16*)carve((size_t)k4H * kIN * 2);   // 8 MB
  __bf16* Whh1b = (__bf16*)carve((size_t)k4H * kH  * 2);   // 8 MB
  __bf16* Wih2b = (__bf16*)carve((size_t)k4H * kH  * 2);   // 8 MB
  __bf16* Whh2b = (__bf16*)carve((size_t)k4H * kH  * 2);   // 8 MB
  __bf16* x_bf  = (__bf16*)carve((size_t)kB  * kIN * 2);   // 8 MB
  __bf16* h1_bf = (__bf16*)carve(BH * 2);                  // 8 MB
  __bf16* h2_bf = (__bf16*)carve(BH * 2);                  // 8 MB
  __bf16* act_bf= (__bf16*)carve(BH * 2);                  // 8 MB
  float* flagcol    = (float*)carve((size_t)k4H * 4);
  float* bias1      = (float*)carve((size_t)k4H * 4);
  float* bias2      = (float*)carve((size_t)k4H * 4);
  float* halt_accum = (float*)carve((size_t)kB * 4);
  float* acc_rem    = (float*)carve((size_t)kB * 4);
  int*   cont       = (int*)  carve((size_t)kB * 4);
  int*   active     = (int*)  carve((size_t)(kT + 1) * 4);
  float* base1      = (float*)carve((size_t)kB * k4H * 4); // 64 MB
  float* gates      = (float*)carve((size_t)kB * k4H * 4); // 64 MB

  const dim3 blk(256);

  // 1) State init (zeros 5*B*H outputs, bf16 h copies, ACT state, active[])
  init_kernel<<<dim3((unsigned)((5 * BH + 255) / 256)), blk, 0, stream>>>(
      outf, h1_bf, h2_bf, halt_accum, acc_rem, cont, active);

  // 2) Weight / input precision prep
  prep_weights_kernel<<<dim3((unsigned)(((size_t)k4H * kIN + 255) / 256)), blk, 0, stream>>>(
      W_ih1, Wih1b, flagcol, b_ih1, b_hh1, bias1, b_ih2, b_hh2, bias2);
  f32_to_bf16_kernel<<<dim3((unsigned)(((size_t)k4H * kH + 255) / 256)), blk, 0, stream>>>(
      W_hh1, Whh1b, (long)k4H * kH);
  f32_to_bf16_kernel<<<dim3((unsigned)(((size_t)k4H * kH + 255) / 256)), blk, 0, stream>>>(
      W_ih2, Wih2b, (long)k4H * kH);
  f32_to_bf16_kernel<<<dim3((unsigned)(((size_t)k4H * kH + 255) / 256)), blk, 0, stream>>>(
      W_hh2, Whh2b, (long)k4H * kH);
  f32_to_bf16_kernel<<<dim3((unsigned)(((size_t)kB * kIN + 255) / 256)), blk, 0, stream>>>(
      inputs, x_bf, (long)kB * kIN);

  // 3) base1 = x @ W_ih1[:, :IN]^T + (b_ih1 + b_hh1)   (step-invariant)
  launch_gemm(base1, x_bf, Wih1b, nullptr, bias1, kB, k4H, kIN, &active[0], stream);

  const unsigned cellBlocks = (unsigned)((BH + 255) / 256);

  // 4) Ponder loop (12 steps, each gated on active[s] != 0 device-side)
  for (int s = 0; s < kT; ++s) {
    const int* act_s = &active[s];
    // Layer 1 gates = base1 + h1 @ W_hh1^T (+ flag column on step 0)
    launch_gemm(gates, h1_bf, Whh1b, base1, (s == 0) ? flagcol : nullptr,
                kB, k4H, kH, act_s, stream);
    lstm_cell_kernel<<<cellBlocks, blk, 0, stream>>>(
        gates, c1, h1, h1_bf, act_bf, act_s);
    // Layer 2 gates = relu(h1) @ W_ih2^T + bias2, then += h2 @ W_hh2^T
    launch_gemm(gates, act_bf, Wih2b, nullptr, bias2, kB, k4H, kH, act_s, stream);
    launch_gemm(gates, h2_bf, Whh2b, gates, nullptr, kB, k4H, kH, act_s, stream);
    lstm_cell_kernel<<<cellBlocks, blk, 0, stream>>>(
        gates, c2, h2, h2_bf, nullptr, act_s);
    // Halting GEMV + ACT accumulation; counts continuers into active[s+1]
    halt_kernel<<<dim3(kB), blk, 0, stream>>>(
        h2, W_halt, b_halt, halt_accum, cont, acc_rem, acc_out, ponder_steps,
        act_s, &active[s + 1]);
  }

  // 5) for-else remainder + ponder cost
  final_acc_kernel<<<cellBlocks, blk, 0, stream>>>(acc_out, h2, cont, halt_accum);
  ponder_cost_kernel<<<dim3(1), blk, 0, stream>>>(acc_rem, ponder_cost);
}
